// HeteroSAGE_24232205484267
// MI455X (gfx1250) — matrix operand order repaired and checked by reading, not compiled, not run on hardware
//
#include <hip/hip_runtime.h>

typedef __attribute__((ext_vector_type(16))) _Float16 v16h;
typedef __attribute__((ext_vector_type(8)))  float    v8f;

#define TB 256
#define APITCH 40    // halves: 80B row stride (20-bank), 16B-aligned fragments
#define WPITCH 136   // halves: 272B col stride, 16B-aligned, K<=128 + pad

// ---------------- zero scratch ----------------
__global__ void zero_kernel(float* __restrict__ p, long n) {
    long i = blockIdx.x * (long)blockDim.x + threadIdx.x;
    if (i < n) p[i] = 0.0f;
}

// ---------------- GEMM: out[M,32] = A[M,K] @ (Wg[+Wg2])[K,32] (+bias0+bias1) -
// 256 threads = 8 wave32 waves; each wave -> 16x32 tile via two
// v_wmma_f32_16x16x32_f16 per 32-wide K step. All in-loop LDS traffic is b128.
__global__ __launch_bounds__(TB)
void gemm_n32_kernel(const float* __restrict__ A,
                     const float* __restrict__ Wg, const float* __restrict__ Wg2,
                     const float* __restrict__ bias0, const float* __restrict__ bias1,
                     float* __restrict__ out, int M, int K) {
    __shared__ __align__(16) _Float16 sWt[32 * WPITCH];   // transposed: [n][k] f16
    __shared__ __align__(16) _Float16 sA[128 * APITCH];   // [r][k] f16
    const int tid  = threadIdx.x;
    const int wave = tid >> 5;
    const int lane = tid & 31;
    const int row0 = blockIdx.x * 128;

    // convert (and optionally sum) weights into LDS, transposed (tiny, once)
    for (int i = tid; i < K * 32; i += TB) {
        int k = i >> 5, n = i & 31;
        float v = Wg[i];
        if (Wg2) v += Wg2[i];
        sWt[n * WPITCH + k] = (_Float16)v;
    }

    // staging assignment: each thread owns half a row (16 contiguous cols)
    const int r   = tid >> 1;              // 0..127
    const int cb  = (tid & 1) << 4;        // 0 | 16
    const int gr  = row0 + r;
    const bool inb = (gr < M);
    const float* __restrict__ arow = A + (size_t)gr * K + cb;

    // wave fragment coordinates
    const int mloc  = (wave << 4) + (lane & 15);
    const int khalf = lane >> 4;
    const int ncol  = lane & 15;

    v8f c0 = {}, c1 = {};
    for (int kk = 0; kk < K; kk += 32) {
        __syncthreads();
        // ---- stage A tile: 4x float4 loads -> 16 halves -> 2x b128 LDS stores
        union { _Float16 h[16]; uint4 q[2]; } pk;
        pk.q[0] = make_uint4(0, 0, 0, 0);
        pk.q[1] = make_uint4(0, 0, 0, 0);
        if (inb) {
            const float4* src = (const float4*)(arow + kk);
            float4 f0 = src[0], f1 = src[1], f2 = src[2], f3 = src[3];
            pk.h[0]  = (_Float16)f0.x; pk.h[1]  = (_Float16)f0.y;
            pk.h[2]  = (_Float16)f0.z; pk.h[3]  = (_Float16)f0.w;
            pk.h[4]  = (_Float16)f1.x; pk.h[5]  = (_Float16)f1.y;
            pk.h[6]  = (_Float16)f1.z; pk.h[7]  = (_Float16)f1.w;
            pk.h[8]  = (_Float16)f2.x; pk.h[9]  = (_Float16)f2.y;
            pk.h[10] = (_Float16)f2.z; pk.h[11] = (_Float16)f2.w;
            pk.h[12] = (_Float16)f3.x; pk.h[13] = (_Float16)f3.y;
            pk.h[14] = (_Float16)f3.z; pk.h[15] = (_Float16)f3.w;
        }
        *(uint4*)&sA[r * APITCH + cb]     = pk.q[0];
        *(uint4*)&sA[r * APITCH + cb + 8] = pk.q[1];
        __syncthreads();

        // ---- A fragment: K = 16*khalf + e -> two b128 loads
        union { v16h v; uint4 q[2]; } a, b0, b1;
        const _Float16* ap = &sA[mloc * APITCH + (khalf << 4)];
        a.q[0] = *(const uint4*)ap;          // e=0..7
        a.q[1] = *(const uint4*)(ap + 8);    // e=8..15
        // ---- B fragment: K = kk + 16*khalf + e, contiguous in sWt column
        const _Float16* bp0 = &sWt[ncol * WPITCH + kk + (khalf << 4)];
        b0.q[0] = *(const uint4*)bp0;
        b0.q[1] = *(const uint4*)(bp0 + 8);
        const _Float16* bp1 = &sWt[(ncol + 16) * WPITCH + kk + (khalf << 4)];
        b1.q[0] = *(const uint4*)bp1;
        b1.q[1] = *(const uint4*)(bp1 + 8);

        c0 = __builtin_amdgcn_wmma_f32_16x16x32_f16(false, a.v, false, b0.v, (short)0, c0, false, false);
        c1 = __builtin_amdgcn_wmma_f32_16x16x32_f16(false, a.v, false, b1.v, (short)0, c1, false, false);
    }

    // C/D layout: lanes 0-15 -> M=r, N=lane; lanes 16-31 -> M=8+r, N=lane-16
    const int mbase = (wave << 4) + (khalf << 3);
    float ba0 = 0.0f, ba1 = 0.0f;
    if (bias0) { ba0 += bias0[ncol]; ba1 += bias0[ncol + 16]; }
    if (bias1) { ba0 += bias1[ncol]; ba1 += bias1[ncol + 16]; }
#pragma unroll
    for (int rr = 0; rr < 8; ++rr) {
        int gm = row0 + mbase + rr;
        if (gm < M) {
            out[(size_t)gm * 32 + ncol]      = c0[rr] + ba0;
            out[(size_t)gm * 32 + 16 + ncol] = c1[rr] + ba1;
        }
    }
}

// ---------------- edge scatter: sum[dst] += t[src], cnt[dst] += 1 ------------
// 32 consecutive lanes handle one edge (coalesced 128B gather + 128B atomic add)
__global__ void scatter_kernel(const float* __restrict__ t, const int* __restrict__ src,
                               const int* __restrict__ dst, int E,
                               float* __restrict__ sum, float* __restrict__ cnt) {
    long idx = blockIdx.x * (long)blockDim.x + threadIdx.x;
    if (idx >= (long)E * 32) return;
    int e = (int)(idx >> 5);
    int h = (int)(idx & 31);
    int s = src[e], d = dst[e];
    atomicAdd(&sum[(size_t)d * 32 + h], t[(size_t)s * 32 + h]);
    if (cnt && h == 0) atomicAdd(&cnt[d], 1.0f);
}

// ---------------- finalize: acc = relu(acc + sum1/max(cnt1,1) [+ sum2/..]) ---
__global__ void finalize_kernel(float* __restrict__ acc,
                                const float* __restrict__ sum1, const float* __restrict__ cnt1,
                                const float* __restrict__ sum2, const float* __restrict__ cnt2,
                                long nNodes) {
    long i = blockIdx.x * (long)blockDim.x + threadIdx.x;
    if (i >= nNodes * 32) return;
    long node = i >> 5;
    float v = acc[i] + sum1[i] / fmaxf(cnt1[node], 1.0f);
    if (sum2) v += sum2[i] / fmaxf(cnt2[node], 1.0f);
    acc[i] = fmaxf(v, 0.0f);
}

// ---------------- readout: out[i] = dot(p2[i,:], Wout) + bout ----------------
// one wave32 per row; shfl_xor tree reduction
__global__ void readout_kernel(const float* __restrict__ acc,
                               const float* __restrict__ sumC, const float* __restrict__ cntC,
                               const float* __restrict__ sumW, const float* __restrict__ cntW,
                               const float* __restrict__ Wout, const float* __restrict__ bout,
                               float* __restrict__ out, int Np) {
    int w    = (int)((blockIdx.x * (long)blockDim.x + threadIdx.x) >> 5);
    int lane = threadIdx.x & 31;
    if (w >= Np) return;
    size_t base = (size_t)w * 32 + lane;
    float v = acc[base] + sumC[base] / fmaxf(cntC[w], 1.0f)
                        + sumW[base] / fmaxf(cntW[w], 1.0f);
    float t = v * Wout[lane];
#pragma unroll
    for (int m = 16; m >= 1; m >>= 1) t += __shfl_xor(t, m, 32);
    if (lane == 0) out[w] = t + bout[0];
}

extern "C" void kernel_launch(void* const* d_in, const int* in_sizes, int n_in,
                              void* d_out, int out_size, void* d_ws, size_t ws_size,
                              hipStream_t stream) {
    const float* x_paper  = (const float*)d_in[0];
    const float* x_author = (const float*)d_in[1];
    const int*   ei_c     = (const int*)d_in[2];
    const int*   ei_w     = (const int*)d_in[3];
    const int*   ei_r     = (const int*)d_in[4];
    const float* W1c_l = (const float*)d_in[5];  const float* b1c = (const float*)d_in[6];
    const float* W1c_r = (const float*)d_in[7];
    const float* W1w_l = (const float*)d_in[8];  const float* b1w = (const float*)d_in[9];
    const float* W1w_r = (const float*)d_in[10];
    const float* W1r_l = (const float*)d_in[11]; const float* b1r = (const float*)d_in[12];
    const float* W1r_r = (const float*)d_in[13];
    const float* W2c_l = (const float*)d_in[14]; const float* b2c = (const float*)d_in[15];
    const float* W2c_r = (const float*)d_in[16];
    const float* W2w_l = (const float*)d_in[17]; const float* b2w = (const float*)d_in[18];
    const float* W2w_r = (const float*)d_in[19];
    // d_in[20..22] = W2r_l, b2r, W2r_r : author output of layer 2 is unused
    const float* W_out = (const float*)d_in[23]; const float* b_out = (const float*)d_in[24];

    const int Dp = 128, Da = 64;
    const int Np = in_sizes[0] / Dp;
    const int Na = in_sizes[1] / Da;
    const int E1 = in_sizes[2] / 2;
    const int E2 = in_sizes[3] / 2;
    const int E3 = in_sizes[4] / 2;

    // ---- workspace layout (all f32) ----
    float* fbase = (float*)d_ws;
    float* tP1  = fbase;                       // [Np,32] t_cites, later t2_cites
    float* tP2  = tP1  + (size_t)Np * 32;      // [Np,32] t_rev,   later p2_acc
    float* tA   = tP2  + (size_t)Np * 32;      // [Na,32] t_writes,later t2_writes
    float* accP = tA   + (size_t)Na * 32;      // [Np,32] paper root acc -> p
    float* accA = accP + (size_t)Np * 32;      // [Na,32] author root acc -> a
    float* sumC = accA + (size_t)Na * 32;      // [Np,32]
    float* sumW = sumC + (size_t)Np * 32;      // [Np,32]
    float* sumR = sumW + (size_t)Np * 32;      // [Na,32]
    float* cntC = sumR + (size_t)Na * 32;      // [Np]
    float* cntW = cntC + Np;                   // [Np]
    float* cntR = cntW + Np;                   // [Na]
    (void)ws_size; (void)n_in; (void)out_size;

    const int tb = TB;
    auto cdiv = [](long a, long b) { return (int)((a + b - 1) / b); };

    // ---- zero all scatter buffers (contiguous region sumC..cntR) ----
    long zlen = (long)Np*32*2 + (long)Na*32 + 2L*Np + Na;
    zero_kernel<<<cdiv(zlen, tb), tb, 0, stream>>>(sumC, zlen);

    // ---- layer 1: pre-transform sources + root terms (WMMA GEMMs) ----
    int gP = cdiv(Np, 128), gA = cdiv(Na, 128);
    gemm_n32_kernel<<<gP, tb, 0, stream>>>(x_paper,  W1c_l, nullptr, nullptr, nullptr, tP1,  Np, 128);
    gemm_n32_kernel<<<gP, tb, 0, stream>>>(x_paper,  W1r_l, nullptr, nullptr, nullptr, tP2,  Np, 128);
    gemm_n32_kernel<<<gA, tb, 0, stream>>>(x_author, W1w_l, nullptr, nullptr, nullptr, tA,   Na,  64);
    gemm_n32_kernel<<<gP, tb, 0, stream>>>(x_paper,  W1c_r, W1w_r,  b1c,     b1w,     accP, Np, 128);
    gemm_n32_kernel<<<gA, tb, 0, stream>>>(x_author, W1r_r, nullptr, b1r,    nullptr, accA, Na,  64);

    // ---- layer 1: edge scatter (mean numerators + degree counts) ----
    scatter_kernel<<<cdiv((long)E1*32, tb), tb, 0, stream>>>(tP1, ei_c, ei_c + E1, E1, sumC, cntC);
    scatter_kernel<<<cdiv((long)E2*32, tb), tb, 0, stream>>>(tA,  ei_w, ei_w + E2, E2, sumW, cntW);
    scatter_kernel<<<cdiv((long)E3*32, tb), tb, 0, stream>>>(tP2, ei_r, ei_r + E3, E3, sumR, cntR);

    // ---- layer 1: finalize (mean + root + ReLU), in place ----
    finalize_kernel<<<cdiv((long)Np*32, tb), tb, 0, stream>>>(accP, sumC, cntC, sumW, cntW, Np);
    finalize_kernel<<<cdiv((long)Na*32, tb), tb, 0, stream>>>(accA, sumR, cntR, nullptr, nullptr, Na);

    // ---- re-zero sums for layer 2 (edge sets identical: counts reused) ----
    zero_kernel<<<cdiv((long)Np*64, tb), tb, 0, stream>>>(sumC, (long)Np*64);

    // ---- layer 2 GEMMs ----
    gemm_n32_kernel<<<gP, tb, 0, stream>>>(accP, W2c_l, nullptr, nullptr, nullptr, tP1, Np, 32);
    gemm_n32_kernel<<<gA, tb, 0, stream>>>(accA, W2w_l, nullptr, nullptr, nullptr, tA,  Na, 32);
    gemm_n32_kernel<<<gP, tb, 0, stream>>>(accP, W2c_r, W2w_r,  b2c,     b2w,     tP2, Np, 32);

    // ---- layer 2 scatter (no counts needed) ----
    scatter_kernel<<<cdiv((long)E1*32, tb), tb, 0, stream>>>(tP1, ei_c, ei_c + E1, E1, sumC, nullptr);
    scatter_kernel<<<cdiv((long)E2*32, tb), tb, 0, stream>>>(tA,  ei_w, ei_w + E2, E2, sumW, nullptr);

    // ---- readout ----
    readout_kernel<<<cdiv((long)Np*32, tb), tb, 0, stream>>>(tP2, sumC, cntC, sumW, cntW,
                                                             W_out, b_out, (float*)d_out, Np);
}
